// MatchLayer_11690900980201
// MI455X (gfx1250) — matrix operand order repaired and checked by reading, not compiled, not run on hardware
//
#include <hip/hip_runtime.h>
#include <cmath>
#include <cstdint>

typedef __attribute__((ext_vector_type(16))) _Float16 v16h;
typedef __attribute__((ext_vector_type(8)))  _Float16 h8;
typedef __attribute__((ext_vector_type(8)))  float    v8f;
typedef __attribute__((ext_vector_type(4)))  unsigned u32x4;
typedef __attribute__((ext_vector_type(8)))  int      i32x8;
typedef __attribute__((ext_vector_type(4)))  int      i32x4;

#define OH 192
#define OW 384
#define PLANE (OH * OW)        // 73728
#define NB 2
#define COUT 32
#define CPAD 224               // max Cin (fusion block: 224)
#define MAXDISP 192

// ---------------------------------------------------------------------------
// TDM: DMA a contiguous slab of `nelem` f16 elements from global to LDS.
// D# built per CDNA5 ISA 8.3/8.4: 1D tile, data_size=2B, tile_dim0=nelem.
// Issued by one wave; caller barriers afterwards.
// ---------------------------------------------------------------------------
__device__ inline void tdm_load_slab(unsigned lds_base, const _Float16* gsrc,
                                     unsigned nelem)
{
    unsigned long long ga = (unsigned long long)(uintptr_t)gsrc;
    u32x4 g0 = { 1u,                                   // count=1, user flags 0
                 lds_base,                             // lds_addr
                 (unsigned)ga,                         // global_addr[31:0]
                 (unsigned)((ga >> 32) & 0x1FFFFFFu) | (2u << 30) }; // type=2
    i32x8 g1 = { (int)(1u << 16),                      // data_size=1 (2 bytes)
                 (int)((nelem & 0xFFFFu) << 16),       // tensor_dim0 lo16
                 (int)((nelem >> 16) | (1u << 16)),    // dim0 hi | tensor_dim1=1
                 (int)(nelem << 16),                   // tile_dim0
                 0,                                    // tile_dim1/2 unused
                 (int)nelem,                           // tensor_dim0_stride lo32
                 0, 0 };
    i32x4 z4 = { 0, 0, 0, 0 };
#if defined(__clang_major__) && (__clang_major__ >= 23)
    i32x8 z8 = {};
    __builtin_amdgcn_tensor_load_to_lds(g0, g1, z4, z4, z8, 0);
#else
    __builtin_amdgcn_tensor_load_to_lds(g0, g1, z4, z4, 0);
#endif
    __builtin_amdgcn_s_wait_tensorcnt(0);
}

// ---------------------------------------------------------------------------
// First-layer implicit-GEMM conv (gather path): f32 NCHW input, Cin=3 (padded
// to K=32), stride 2. Output raw f32 NCHW [2,32,192,384]. 4 waves / block,
// each wave: 16 spatial x 32 channels via two v_wmma_f32_16x16x32_f16.
// ---------------------------------------------------------------------------
template <int KH, int KW>
__global__ __launch_bounds__(128)
void conv_gather_kernel(const float* __restrict__ in, int Cin,
                        int IH, int IW, int stride, int pad,
                        const float* __restrict__ Wt,
                        float* __restrict__ out)
{
    const int KcC = ((Cin + 31) >> 5) << 5;
    __shared__ _Float16 wlds[COUT * 32];

    const int m_base = blockIdx.x * 64;
    const int row = m_base / OW;
    const int ox0 = m_base % OW;
    const int n   = row / OH;
    const int oy  = row % OH;

    const int wave = threadIdx.x >> 5;
    const int lane = threadIdx.x & 31;
    const int mlo  = lane & 15;
    const int hi   = lane >> 4;
    const int ox_base = ox0 + wave * 16;
    const int ax      = ox_base + mlo;

    v8f acc0 = {};
    v8f acc1 = {};

    for (int ky = 0; ky < KH; ++ky) {
        for (int kx = 0; kx < KW; ++kx) {
            __syncthreads();
            for (int i = threadIdx.x; i < COUT * KcC; i += 128) {
                const int co = i / KcC;
                const int c  = i - co * KcC;
                const float w = (c < Cin)
                    ? Wt[((size_t)(co * Cin + c) * KH + ky) * KW + kx] : 0.0f;
                wlds[co * 32 + c] = (_Float16)w;
            }
            __syncthreads();

            const int iy = oy * stride + ky - pad;
            const int ix = ax * stride + kx - pad;
            const bool inb = (iy >= 0) & (iy < IH) & (ix >= 0) & (ix < IW);
            if (inb)
                __builtin_prefetch(in + ((size_t)(n * Cin) * IH + iy) * IW + ix, 0, 1);

            v16h a;
            #pragma unroll
            for (int h = 0; h < 16; ++h) {
                const int k = (h & 7) + ((h >> 3) << 4) + (hi << 3);
                float v = 0.0f;
                if (inb && k < Cin)
                    v = in[((size_t)(n * Cin + k) * IH + iy) * IW + ix];
                a[h] = (_Float16)v;
            }
            v16h b0, b1;
            #pragma unroll
            for (int h = 0; h < 16; ++h) {
                const int c = (h & 7) + ((h >> 3) << 4) + (hi << 3);
                b0[h] = wlds[mlo * 32 + c];
                b1[h] = wlds[(mlo + 16) * 32 + c];
            }
            acc0 = __builtin_amdgcn_wmma_f32_16x16x32_f16(
                       false, a, false, b0, (short)0, acc0, false, false);
            acc1 = __builtin_amdgcn_wmma_f32_16x16x32_f16(
                       false, a, false, b1, (short)0, acc1, false, false);
        }
    }

    #pragma unroll
    for (int r = 0; r < 8; ++r) {
        const int m  = r + (hi << 3);
        const int ox = ox_base + m;
        out[((size_t)(n * COUT + mlo)      * OH + oy) * OW + ox] = acc0[r];
        out[((size_t)(n * COUT + mlo + 16) * OH + oy) * OW + ox] = acc1[r];
    }
}

// ---------------------------------------------------------------------------
// Main implicit-GEMM conv on f16 NHWC activations (stride 1, 192x384).
// Virtual concat input: channels [0,C0) from s0, [C0,C0+C1) from s1;
// C0/C1 multiples of 32 so each 32-wide K chunk stays in one source.
// Weights pre-packed f16 [tap][co][c]; per-tap slab DMA'd to LDS by TDM.
// Output raw f32 NCHW (feeds BN statistics).
// ---------------------------------------------------------------------------
template <int KH, int KW>
__global__ __launch_bounds__(128)
void conv_wmma_nhwc(const _Float16* __restrict__ s0, int C0,
                    const _Float16* __restrict__ s1, int C1,
                    const _Float16* __restrict__ wpack,
                    float* __restrict__ out)
{
    const int Cin = C0 + C1;
    constexpr int pad = (KH == 3) ? 1 : 0;
    __shared__ _Float16 wlds[COUT * CPAD];

    const int m_base = blockIdx.x * 64;
    const int row = m_base / OW;
    const int ox0 = m_base % OW;
    const int n   = row / OH;
    const int oy  = row % OH;

    const int wave = threadIdx.x >> 5;
    const int lane = threadIdx.x & 31;
    const int mlo  = lane & 15;
    const int hi   = lane >> 4;
    const int ox_base = ox0 + wave * 16;
    const int ax      = ox_base + mlo;

    const unsigned slab = (unsigned)COUT * (unsigned)Cin;
    const unsigned lds_base = (unsigned)(uintptr_t)&wlds[0];

    v8f acc0 = {};
    v8f acc1 = {};

    for (int ky = 0; ky < KH; ++ky) {
        for (int kx = 0; kx < KW; ++kx) {
            const int tap = ky * KW + kx;
            __syncthreads();
            if (wave == 0)   // one wave issues the tensor DMA, waits on TENSORcnt
                tdm_load_slab(lds_base, wpack + (size_t)tap * slab, slab);
            __syncthreads();

            const int iy = oy + ky - pad;
            const int ix = ax + kx - pad;
            const bool inb = (iy >= 0) & (iy < OH) & (ix >= 0) & (ix < OW);
            const size_t pix = (size_t)(n * OH + iy) * OW + ix;

            for (int kc = 0; kc < Cin; kc += 32) {
                // A fragment: two 16-byte vector loads from NHWC f16
                v16h a = {};
                if (inb) {
                    const _Float16* src = (kc < C0)
                        ? s0 + pix * C0 + kc
                        : s1 + pix * C1 + (kc - C0);
                    const h8* p = (const h8*)src;
                    const h8 lo = p[hi];       // k = hi*8 + h
                    const h8 hh = p[2 + hi];   // k = 16 + hi*8 + h
                    #pragma unroll
                    for (int h = 0; h < 8; ++h) { a[h] = lo[h]; a[8 + h] = hh[h]; }
                }
                // B fragments from LDS slab [co][c]
                v16h b0, b1;
                #pragma unroll
                for (int h = 0; h < 16; ++h) {
                    const int c = kc + (h & 7) + ((h >> 3) << 4) + (hi << 3);
                    b0[h] = wlds[mlo * Cin + c];
                    b1[h] = wlds[(mlo + 16) * Cin + c];
                }
                acc0 = __builtin_amdgcn_wmma_f32_16x16x32_f16(
                           false, a, false, b0, (short)0, acc0, false, false);
                acc1 = __builtin_amdgcn_wmma_f32_16x16x32_f16(
                           false, a, false, b1, (short)0, acc1, false, false);
            }
        }
    }

    #pragma unroll
    for (int r = 0; r < 8; ++r) {
        const int m  = r + (hi << 3);
        const int ox = ox_base + m;
        out[((size_t)(n * COUT + mlo)      * OH + oy) * OW + ox] = acc0[r];
        out[((size_t)(n * COUT + mlo + 16) * OH + oy) * OW + ox] = acc1[r];
    }
}

// Pack f32 OIHW weights -> f16 [tap][co][c]
__global__ void prep_weights_kernel(const float* __restrict__ Wt,
                                    _Float16* __restrict__ wpack,
                                    int Cin, int taps)
{
    const int total = taps * COUT * Cin;
    for (int i = blockIdx.x * blockDim.x + threadIdx.x; i < total;
         i += gridDim.x * blockDim.x) {
        const int t  = i / (COUT * Cin);
        const int r  = i - t * (COUT * Cin);
        const int co = r / Cin;
        const int c  = r - co * Cin;
        wpack[i] = (_Float16)Wt[(size_t)(co * Cin + c) * taps + t];
    }
}

// ---------------------------------------------------------------------------
// BN statistics over raw f32 NCHW conv output [2,32,PLANE].
// ---------------------------------------------------------------------------
__global__ void stats_kernel(const float* __restrict__ x, float* __restrict__ stats)
{
    const int ch = blockIdx.x;
    __shared__ float ssum[256];
    __shared__ float ssq[256];
    float s = 0.0f, q = 0.0f;
    const int total = NB * PLANE;
    for (int i = blockIdx.y * blockDim.x + threadIdx.x; i < total;
         i += gridDim.y * blockDim.x) {
        const int nn  = i / PLANE;
        const int off = i - nn * PLANE;
        const float v = x[((size_t)(nn * COUT + ch)) * PLANE + off];
        s += v; q += v * v;
    }
    ssum[threadIdx.x] = s; ssq[threadIdx.x] = q;
    __syncthreads();
    for (int st = 128; st > 0; st >>= 1) {
        if ((int)threadIdx.x < st) {
            ssum[threadIdx.x] += ssum[threadIdx.x + st];
            ssq[threadIdx.x]  += ssq[threadIdx.x + st];
        }
        __syncthreads();
    }
    if (threadIdx.x == 0) {
        atomicAdd(&stats[ch], ssum[0]);
        atomicAdd(&stats[32 + ch], ssq[0]);
    }
}

__global__ void finalize_bn_kernel(const float* __restrict__ stats,
                                   const float* __restrict__ g,
                                   const float* __restrict__ b,
                                   float inv_cnt, float* __restrict__ sb)
{
    const int c = threadIdx.x;
    if (c < 32) {
        const float mean = stats[c] * inv_cnt;
        const float var  = stats[32 + c] * inv_cnt - mean * mean;
        const float sc   = g[c] * rsqrtf(var + 1e-5f);
        sb[c]      = sc;
        sb[32 + c] = b[c] - mean * sc;
    }
}

// ---------------------------------------------------------------------------
// BN epilogues. Raw conv data is f32 NCHW; conv-feeding outputs are f16 NHWC.
// ---------------------------------------------------------------------------
__global__ void bn_relu_nhwc_kernel(const float* __restrict__ x,
                                    const float* __restrict__ sb,
                                    _Float16* __restrict__ y, int total)
{
    for (int i = blockIdx.x * blockDim.x + threadIdx.x; i < total;
         i += gridDim.x * blockDim.x) {
        const int n   = i / (COUT * PLANE);
        const int r   = i - n * (COUT * PLANE);
        const int c   = r / PLANE;
        const int off = r - c * PLANE;
        float v = fmaf(x[i], sb[c], sb[32 + c]);
        v = v > 0.0f ? v : 0.0f;
        y[((size_t)n * PLANE + off) * COUT + c] = (_Float16)v;
    }
}

// y_nhwc = relu(bn1(x1) + bn2(x2))
__global__ void bn2_add_relu_nhwc_kernel(const float* __restrict__ x1,
                                         const float* __restrict__ sb1,
                                         const float* __restrict__ x2,
                                         const float* __restrict__ sb2,
                                         _Float16* __restrict__ y, int total)
{
    for (int i = blockIdx.x * blockDim.x + threadIdx.x; i < total;
         i += gridDim.x * blockDim.x) {
        const int n   = i / (COUT * PLANE);
        const int r   = i - n * (COUT * PLANE);
        const int c   = r / PLANE;
        const int off = r - c * PLANE;
        float v = fmaf(x1[i], sb1[c], sb1[32 + c]) +
                  fmaf(x2[i], sb2[c], sb2[32 + c]);
        v = v > 0.0f ? v : 0.0f;
        y[((size_t)n * PLANE + off) * COUT + c] = (_Float16)v;
    }
}

// block output: yf32 (NCHW, d_out) and yh (f16 NHWC copy); res is f16 NHWC
__global__ void bn_add_relu_final_kernel(const float* __restrict__ x,
                                         const float* __restrict__ sb,
                                         const _Float16* __restrict__ res,
                                         float* __restrict__ yf,
                                         _Float16* __restrict__ yh, int total)
{
    for (int i = blockIdx.x * blockDim.x + threadIdx.x; i < total;
         i += gridDim.x * blockDim.x) {
        const int n   = i / (COUT * PLANE);
        const int r   = i - n * (COUT * PLANE);
        const int c   = r / PLANE;
        const int off = r - c * PLANE;
        const size_t hidx = ((size_t)n * PLANE + off) * COUT + c;
        float v = fmaf(x[i], sb[c], sb[32 + c]) + (float)res[hidx];
        v = v > 0.0f ? v : 0.0f;
        yf[i] = v;
        yh[hidx] = (_Float16)v;
    }
}

// final fusion output: f32 NCHW = relu(bn1(x1)+bn2(x2))
__global__ void bn2_add_relu_f32_kernel(const float* __restrict__ x1,
                                        const float* __restrict__ sb1,
                                        const float* __restrict__ x2,
                                        const float* __restrict__ sb2,
                                        float* __restrict__ y, int total)
{
    for (int i = blockIdx.x * blockDim.x + threadIdx.x; i < total;
         i += gridDim.x * blockDim.x) {
        const int c = (i / PLANE) & 31;
        const float v = fmaf(x1[i], sb1[c], sb1[32 + c]) +
                        fmaf(x2[i], sb2[c], sb2[32 + c]);
        y[i] = v > 0.0f ? v : 0.0f;
    }
}

// ---------------------------------------------------------------------------
// Correlation volume, NHWC f16 in/out:
// corr[b,h,w,d] = lrelu( (1/32) sum_c L[b,h,w,c]*R[b,h,w-d,c] ), 0 if w<d.
// ---------------------------------------------------------------------------
__global__ __launch_bounds__(256)
void corr_nhwc_kernel(const _Float16* __restrict__ L,
                      const _Float16* __restrict__ R,
                      _Float16* __restrict__ out)
{
    const int b  = blockIdx.z;
    const int h  = blockIdx.y;
    const int w0 = blockIdx.x * 64;

    __shared__ _Float16 ls[64][32];
    __shared__ _Float16 rs[256][32];   // covers w0-192 .. w0+63

    for (int i = threadIdx.x; i < 64 * 32; i += 256) {
        const int w = i >> 5, c = i & 31;
        ls[w][c] = L[((size_t)(b * OH + h) * OW + w0 + w) * COUT + c];
    }
    for (int i = threadIdx.x; i < 256 * 32; i += 256) {
        const int w = i >> 5, c = i & 31;
        const int gw = w0 - MAXDISP + w;
        rs[w][c] = (gw >= 0)
            ? R[((size_t)(b * OH + h) * OW + gw) * COUT + c] : (_Float16)0.0f;
    }
    __syncthreads();

    const int w  = threadIdx.x & 63;
    const int dg = threadIdx.x >> 6;
    const size_t obase = ((size_t)(b * OH + h) * OW + w0 + w) * MAXDISP;
    for (int d = dg; d < MAXDISP; d += 4) {
        const int rw = w + MAXDISP - d;
        float s = 0.0f;
        #pragma unroll
        for (int c = 0; c < 32; ++c)
            s = fmaf((float)ls[w][c], (float)rs[rw][c], s);
        s *= (1.0f / 32.0f);
        s = s > 0.0f ? s : 0.1f * s;
        out[obase + d] = (_Float16)s;
    }
}

// ---------------------------------------------------------------------------
// Host orchestration
// ---------------------------------------------------------------------------
extern "C" void kernel_launch(void* const* d_in, const int* in_sizes, int n_in,
                              void* d_out, int out_size, void* d_ws, size_t ws_size,
                              hipStream_t stream)
{
    (void)in_sizes; (void)n_in; (void)out_size; (void)ws_size;

    const float* LFT = (const float*)d_in[0];
    const float* RGT = (const float*)d_in[1];
    const float* b1_w1 = (const float*)d_in[2];
    const float* b1_w2 = (const float*)d_in[3];
    const float* b1_ws = (const float*)d_in[4];
    const float* b2_w1 = (const float*)d_in[5];
    const float* b2_w2 = (const float*)d_in[6];
    const float* fb_w1 = (const float*)d_in[7];
    const float* fb_w2 = (const float*)d_in[8];
    const float* fb_ws = (const float*)d_in[9];
    const float* b1_g1 = (const float*)d_in[10]; const float* b1_b1 = (const float*)d_in[11];
    const float* b1_g2 = (const float*)d_in[12]; const float* b1_b2 = (const float*)d_in[13];
    const float* b1_gs = (const float*)d_in[14]; const float* b1_bs = (const float*)d_in[15];
    const float* b2_g1 = (const float*)d_in[16]; const float* b2_b1 = (const float*)d_in[17];
    const float* b2_g2 = (const float*)d_in[18]; const float* b2_b2 = (const float*)d_in[19];
    const float* fb_g1 = (const float*)d_in[20]; const float* fb_b1 = (const float*)d_in[21];
    const float* fb_g2 = (const float*)d_in[22]; const float* fb_b2 = (const float*)d_in[23];
    const float* fb_gs = (const float*)d_in[24]; const float* fb_bs = (const float*)d_in[25];

    const size_t TEN   = (size_t)NB * COUT * PLANE;        // 4,718,592
    const size_t CORRN = (size_t)NB * MAXDISP * PLANE;     // 28,311,552

    char* ws = (char*)d_ws;
    float*    A     = (float*)ws;           ws += TEN * 4;     // raw conv (f32 NCHW)
    float*    Csc   = (float*)ws;           ws += TEN * 4;     // raw shortcut
    _Float16* ACT0  = (_Float16*)ws;        ws += TEN * 2;     // f16 NHWC temps
    _Float16* ACT1  = (_Float16*)ws;        ws += TEN * 2;
    _Float16* ACT2  = (_Float16*)ws;        ws += TEN * 2;
    _Float16* BLh   = (_Float16*)ws;        ws += TEN * 2;     // block_l f16 NHWC
    _Float16* BRh   = (_Float16*)ws;        ws += TEN * 2;     // block_r f16 NHWC
    _Float16* CORRh = (_Float16*)ws;        ws += CORRN * 2;   // corr f16 NHWC
    _Float16* WPACK = (_Float16*)ws;        ws += 65536 * 2;   // packed weights
    float*    STATS = (float*)ws;           ws += 64 * 4;
    float*    SB1   = (float*)ws;           ws += 64 * 4;
    float*    SB2   = (float*)ws;

    float* OUT_L = (float*)d_out;
    float* OUT_R = OUT_L + TEN;
    float* OUT_C = OUT_L + 2 * TEN;

    const int CONV_GRID = (NB * OH * OW) / 64;             // 2304
    const int EW_GRID   = 2048;
    const int TOT       = (int)TEN;
    const float INV_CNT = 1.0f / (float)(NB * PLANE);

    auto bnstats = [&](const float* x, const float* g, const float* b, float* sb) {
        hipMemsetAsync(STATS, 0, 64 * sizeof(float), stream);
        stats_kernel<<<dim3(32, 8), 256, 0, stream>>>(x, STATS);
        finalize_bn_kernel<<<1, 32, 0, stream>>>(STATS, g, b, INV_CNT, sb);
    };
    auto convN3 = [&](const _Float16* s0, int c0, const _Float16* s1, int c1,
                      const float* w, float* o) {
        prep_weights_kernel<<<128, 256, 0, stream>>>(w, WPACK, c0 + c1, 9);
        conv_wmma_nhwc<3, 3><<<CONV_GRID, 128, 0, stream>>>(s0, c0, s1, c1, WPACK, o);
    };
    auto convN1 = [&](const _Float16* s0, int c0, const _Float16* s1, int c1,
                      const float* w, float* o) {
        prep_weights_kernel<<<128, 256, 0, stream>>>(w, WPACK, c0 + c1, 1);
        conv_wmma_nhwc<1, 1><<<CONV_GRID, 128, 0, stream>>>(s0, c0, s1, c1, WPACK, o);
    };

    auto side = [&](const float* inp, float* blk_f32, _Float16* blk_h) {
        // block1 conv1 (3ch, stride2) -> bn+relu -> NHWC f16
        conv_gather_kernel<3, 3><<<CONV_GRID, 128, 0, stream>>>(
            inp, 3, 384, 768, 2, 1, b1_w1, A);
        bnstats(A, b1_g1, b1_b1, SB1);
        bn_relu_nhwc_kernel<<<EW_GRID, 256, 0, stream>>>(A, SB1, ACT0, TOT);
        // block1 conv2 + 1x1 shortcut
        convN3(ACT0, 32, nullptr, 0, b1_w2, A);
        bnstats(A, b1_g2, b1_b2, SB1);
        conv_gather_kernel<1, 1><<<CONV_GRID, 128, 0, stream>>>(
            inp, 3, 384, 768, 2, 0, b1_ws, Csc);
        bnstats(Csc, b1_gs, b1_bs, SB2);
        bn2_add_relu_nhwc_kernel<<<EW_GRID, 256, 0, stream>>>(A, SB1, Csc, SB2,
                                                              ACT1, TOT);
        // block2 (identity shortcut = ACT1)
        convN3(ACT1, 32, nullptr, 0, b2_w1, A);
        bnstats(A, b2_g1, b2_b1, SB1);
        bn_relu_nhwc_kernel<<<EW_GRID, 256, 0, stream>>>(A, SB1, ACT2, TOT);
        convN3(ACT2, 32, nullptr, 0, b2_w2, A);
        bnstats(A, b2_g2, b2_b2, SB1);
        bn_add_relu_final_kernel<<<EW_GRID, 256, 0, stream>>>(A, SB1, ACT1,
                                                              blk_f32, blk_h, TOT);
    };

    side(LFT, OUT_L, BLh);
    side(RGT, OUT_R, BRh);

    corr_nhwc_kernel<<<dim3(OW / 64, OH, NB), 256, 0, stream>>>(BLh, BRh, CORRh);

    // fusion block over virtual concat [block_l(32) | corr(192)]
    convN3(BLh, 32, CORRh, MAXDISP, fb_w1, A);
    bnstats(A, fb_g1, fb_b1, SB1);
    bn_relu_nhwc_kernel<<<EW_GRID, 256, 0, stream>>>(A, SB1, ACT0, TOT);
    convN3(ACT0, 32, nullptr, 0, fb_w2, A);
    bnstats(A, fb_g2, fb_b2, SB1);
    convN1(BLh, 32, CORRh, MAXDISP, fb_ws, Csc);
    bnstats(Csc, fb_gs, fb_bs, SB2);
    bn2_add_relu_f32_kernel<<<EW_GRID, 256, 0, stream>>>(A, SB1, Csc, SB2,
                                                         OUT_C, TOT);
}
